// GATEncoder_39565238731351
// MI455X (gfx1250) — matrix operand order repaired and checked by reading, not compiled, not run on hardware
//
#include <hip/hip_runtime.h>

// ---------------------------------------------------------------------------
// GAT encoder for gfx1250 (MI455X).
//  - Dense projections: v_wmma_f32_16x16x32_bf16, A/B staged in LDS with
//    layouts matching the ISA fragment maps so every fragment is built from
//    two ds_load_b128 ops (no scalar u16 gathers).
//  - W tile staged via GLOBAL_LOAD_ASYNC_TO_LDS_B128 (ASYNCcnt) when the
//    toolchain exposes the builtin; synchronous fallback otherwise.
//  - Edge softmax: u32-monotonic-key atomic max + f32 atomic adds.
//  - Aggregation: float4 gather of h[src], 4x global_atomic_add_f32 scatter.
// All launches on `stream`; scratch only in d_ws (graph-capture safe).
// ---------------------------------------------------------------------------

typedef __attribute__((ext_vector_type(16))) __bf16 v16bf;
typedef __attribute__((ext_vector_type(8)))  __bf16 v8bf;
typedef __attribute__((ext_vector_type(4)))  __bf16 v4bf;
typedef __attribute__((ext_vector_type(8)))  float  v8f;
typedef __attribute__((ext_vector_type(4)))  int    v4i;

// pointer-to-int4 in global (AS1) / LDS (AS3) address spaces for async copies
typedef v4i __attribute__((address_space(1)))* g_v4i_p;
typedef v4i __attribute__((address_space(3)))* l_v4i_p;

#define N_  50000
#define E_  800000
#define D_  128      // = H*C
#define H_  8
#define C_  16
#define L_  3
#define G_  100
#define LAT_ 64
#define EP_ (E_ + N_)          // edges + self loops = 850000
#define NEG_SLOPE 0.2f
#define BN_EPS 1e-5f

#define ROWS_ 80               // rows per GEMM block (50000 = 625 * 80)
#define SUB_  5                // 16-row subtiles per block

#if __has_builtin(__builtin_amdgcn_global_load_async_to_lds_b128)
#define USE_ASYNC_LDS 1
#endif

// ---- monotonic float <-> uint mapping for atomicMax on floats --------------
__device__ __forceinline__ unsigned f2key(float f) {
  unsigned u = __float_as_uint(f);
  return (u & 0x80000000u) ? ~u : (u | 0x80000000u);
}
__device__ __forceinline__ float key2f(unsigned k) {
  unsigned u = (k & 0x80000000u) ? (k & 0x7FFFFFFFu) : ~k;
  return __uint_as_float(u);
}

__device__ __forceinline__ void get_edge(const int* __restrict__ ei, int idx,
                                         int& s, int& d) {
  if (idx < E_) { s = ei[idx]; d = ei[E_ + idx]; }
  else          { s = idx - E_; d = s; }
}

// ---- one-time W -> bf16 transposed ( Wt[l][n][k] = W[l][k][n] ) ------------
__global__ void k_w_to_bf16t(const float* __restrict__ W,
                             __bf16* __restrict__ Wt) {
  int i = blockIdx.x * blockDim.x + threadIdx.x;
  if (i >= L_ * D_ * D_) return;
  const int l = i >> 14, r = i & 16383, k = r >> 7, n = r & 127;
  Wt[l * 16384 + n * 128 + k] = (__bf16)W[i];
}

// ---------------------------------------------------------------------------
// GEMM: Y[N,128] = X[N,128] @ W[128,128]  via v_wmma_f32_16x16x32_bf16.
// Block = 256 thr = 8 waves, owns 80 rows; wave w owns column tile w (16 cols)
// for all 5 row subtiles.  K=128 -> 4 WMMA steps; each B fragment reused by
// 5 A fragments (20 WMMA / wave).
// Fragment reads (ISA 7.12.2 layouts):
//   A (16x32 bf16): lane m=l&15, K-chunks {kbA+0..7, kbA+16..23}, kbA=8*(l>>4)
//       -> two 16B contiguous runs in As[m][k]           (2x ds_load_b128)
//   B (32x16 bf16): lane n=l&15, K-chunks {kbB+0..7, kbB+16..23}, kbB=8*(l>>4)
//       -> two 16B contiguous runs in Bt[n][k]           (2x ds_load_b128)
// ---------------------------------------------------------------------------
__global__ __launch_bounds__(256) void k_gemm_wmma(
    const float* __restrict__ X, const __bf16* __restrict__ Wt,
    float* __restrict__ Y) {
  __shared__ alignas(16) __bf16 As[ROWS_][D_ + 8];
  __shared__ alignas(16) __bf16 Bt[D_][D_ + 8];

  const int t = threadIdx.x;
  // stage transposed bf16 W: 2048 chunks of 8 bf16 (16B each)
#ifdef USE_ASYNC_LDS
  for (int c = t; c < D_ * 16; c += 256) {
    const int n = c >> 4, kc = (c & 15) * 8;
    __builtin_amdgcn_global_load_async_to_lds_b128(
        (g_v4i_p)(uintptr_t)(Wt + n * 128 + kc),
        (l_v4i_p)(unsigned)(uintptr_t)(&Bt[n][kc]),
        0, 0);
  }
#else
  for (int c = t; c < D_ * 16; c += 256) {
    const int n = c >> 4, kc = (c & 15) * 8;
    *(v8bf*)&Bt[n][kc] = *(const v8bf*)&Wt[n * 128 + kc];
  }
#endif
  // stage 80 activation rows: float4 load -> 4x bf16 pack -> ds_store_b64
  const int row0 = blockIdx.x * ROWS_;
  for (int c = t; c < ROWS_ * 32; c += 256) {
    const int r = c >> 5, kc = (c & 31) * 4;
    const float4 f = *(const float4*)&X[(size_t)(row0 + r) * D_ + kc];
    v4bf p;
    p[0] = (__bf16)f.x; p[1] = (__bf16)f.y;
    p[2] = (__bf16)f.z; p[3] = (__bf16)f.w;
    *(v4bf*)&As[r][kc] = p;
  }
#ifdef USE_ASYNC_LDS
#if __has_builtin(__builtin_amdgcn_s_wait_asynccnt)
  __builtin_amdgcn_s_wait_asynccnt(0);
#else
  asm volatile("s_wait_asynccnt 0" ::: "memory");
#endif
#endif
  __syncthreads();

  const int wave = t >> 5;
  const int lane = t & 31;
  const int nloc = lane & 15;
  const int hi   = (lane >> 4) & 1;
  const int ncol = wave * 16 + nloc;

  v8f acc[SUB_] = {};
#pragma unroll
  for (int kt = 0; kt < 4; ++kt) {
    const int k0 = kt * 32;
    const int kbB = k0 + (hi ? 8 : 0);
    const v8bf b0 = *(const v8bf*)&Bt[ncol][kbB];
    const v8bf b1 = *(const v8bf*)&Bt[ncol][kbB + 16];
    const v16bf b = __builtin_shufflevector(b0, b1, 0, 1, 2, 3, 4, 5, 6, 7,
                                            8, 9, 10, 11, 12, 13, 14, 15);
    const int kbA = k0 + (hi ? 8 : 0);
#pragma unroll
    for (int s = 0; s < SUB_; ++s) {
      const int m = s * 16 + nloc;
      const v8bf a0 = *(const v8bf*)&As[m][kbA];
      const v8bf a1 = *(const v8bf*)&As[m][kbA + 16];
      const v16bf a = __builtin_shufflevector(a0, a1, 0, 1, 2, 3, 4, 5, 6, 7,
                                              8, 9, 10, 11, 12, 13, 14, 15);
      acc[s] = __builtin_amdgcn_wmma_f32_16x16x32_bf16(
          false, a, false, b, (short)0, acc[s], false, false);
    }
  }
  // D layout: VGPR v -> row M = v + 8*hi, N striped over lanes 0..15
#pragma unroll
  for (int s = 0; s < SUB_; ++s) {
#pragma unroll
    for (int v = 0; v < 8; ++v) {
      const int m = s * 16 + v + (hi ? 8 : 0);
      Y[(size_t)(row0 + m) * D_ + ncol] = acc[s][v];
    }
  }
}

// ---- per-node attention logits: alpha[n,h] = sum_c h[n,h,c] * a[h,c] -------
__global__ void k_node_alpha(const float* __restrict__ Hf,
                             const float* __restrict__ aS,
                             const float* __restrict__ aD,
                             float* __restrict__ oS, float* __restrict__ oD) {
  int i = blockIdx.x * blockDim.x + threadIdx.x;
  if (i >= N_ * H_) return;
  const int n = i >> 3, h = i & 7;
  float s = 0.f, d = 0.f;
#pragma unroll
  for (int q = 0; q < 4; ++q) {
    const float4 hv = *(const float4*)&Hf[(size_t)n * D_ + h * C_ + q * 4];
    const float4 sv = *(const float4*)&aS[h * C_ + q * 4];
    const float4 dv = *(const float4*)&aD[h * C_ + q * 4];
    s += hv.x * sv.x + hv.y * sv.y + hv.z * sv.z + hv.w * sv.w;
    d += hv.x * dv.x + hv.y * dv.y + hv.z * dv.z + hv.w * dv.w;
  }
  oS[i] = s;
  oD[i] = d;
}

__global__ void k_seg_init(unsigned* __restrict__ segmax,
                           float* __restrict__ segsum) {
  int i = blockIdx.x * blockDim.x + threadIdx.x;
  if (i < N_ * H_) { segmax[i] = 0u; segsum[i] = 0.f; }
}

__device__ __forceinline__ float edge_logit(const float* aS, const float* aD,
                                            int s, int d, int h) {
  const float v = aS[s * H_ + h] + aD[d * H_ + h];
  return v > 0.f ? v : NEG_SLOPE * v;
}

__global__ void k_edge_max(const int* __restrict__ ei,
                           const float* __restrict__ aS,
                           const float* __restrict__ aD,
                           unsigned* __restrict__ segmax) {
  int i = blockIdx.x * blockDim.x + threadIdx.x;
  if (i >= EP_ * H_) return;
  const int e = i >> 3, h = i & 7;
  int s, d; get_edge(ei, e, s, d);
  atomicMax(&segmax[d * H_ + h], f2key(edge_logit(aS, aD, s, d, h)));
}

__global__ void k_edge_sum(const int* __restrict__ ei,
                           const float* __restrict__ aS,
                           const float* __restrict__ aD,
                           const unsigned* __restrict__ segmax,
                           float* __restrict__ segsum) {
  int i = blockIdx.x * blockDim.x + threadIdx.x;
  if (i >= EP_ * H_) return;
  const int e = i >> 3, h = i & 7;
  int s, d; get_edge(ei, e, s, d);
  const float m = key2f(segmax[d * H_ + h]);
  atomicAdd(&segsum[d * H_ + h], expf(edge_logit(aS, aD, s, d, h) - m));
}

__global__ void k_edge_alpha(const int* __restrict__ ei,
                             const float* __restrict__ aS,
                             const float* __restrict__ aD,
                             const unsigned* __restrict__ segmax,
                             const float* __restrict__ segsum,
                             float* __restrict__ alphaE) {
  int i = blockIdx.x * blockDim.x + threadIdx.x;
  if (i >= EP_ * H_) return;
  const int e = i >> 3, h = i & 7;
  int s, d; get_edge(ei, e, s, d);
  const float m  = key2f(segmax[d * H_ + h]);
  const float ex = expf(edge_logit(aS, aD, s, d, h) - m);
  alphaE[i] = ex / (segsum[d * H_ + h] + 1e-16f);
}

__global__ void k_out_init(float4* __restrict__ out,
                           const float4* __restrict__ bias4) {
  int i = blockIdx.x * blockDim.x + threadIdx.x;
  if (i < N_ * 32) out[i] = bias4[i & 31];
}

// 4 channels per thread: one float4 gather of h[src], 4 atomic f32 scatters.
__global__ void k_aggregate(const int* __restrict__ ei,
                            const float* __restrict__ proj,
                            const float* __restrict__ alphaE,
                            float* __restrict__ out) {
  int i = blockIdx.x * blockDim.x + threadIdx.x;
  if (i >= EP_ * 32) return;
  const int e = i >> 5, q = i & 31;
  const int c0 = q * 4, h = q >> 2;
  int s, d; get_edge(ei, e, s, d);
  const float a = alphaE[e * H_ + h];
  const float4 p = *(const float4*)&proj[(size_t)s * D_ + c0];
  float* o = &out[(size_t)d * D_ + c0];
  atomicAdd(o + 0, p.x * a);
  atomicAdd(o + 1, p.y * a);
  atomicAdd(o + 2, p.z * a);
  atomicAdd(o + 3, p.w * a);
}

__global__ void k_elu(float4* __restrict__ x) {
  int i = blockIdx.x * blockDim.x + threadIdx.x;
  if (i >= N_ * 32) return;
  float4 v = x[i];
  v.x = v.x > 0.f ? v.x : (expf(v.x) - 1.f);
  v.y = v.y > 0.f ? v.y : (expf(v.y) - 1.f);
  v.z = v.z > 0.f ? v.z : (expf(v.z) - 1.f);
  v.w = v.w > 0.f ? v.w : (expf(v.w) - 1.f);
  x[i] = v;
}

// ---- pooling / BN / FC -----------------------------------------------------
__global__ void k_pool_init(float* __restrict__ pooled) {
  int i = blockIdx.x * blockDim.x + threadIdx.x;
  if (i < G_ * D_) pooled[i] = 0.f;
}

__global__ void k_pool(const float* __restrict__ Hf,
                       const int* __restrict__ batch,
                       float* __restrict__ pooled) {
  int i = blockIdx.x * blockDim.x + threadIdx.x;
  if (i >= N_ * 32) return;
  const int n = i >> 5, c0 = (i & 31) * 4;
  const float4 v = *(const float4*)&Hf[(size_t)n * D_ + c0];
  float* p = &pooled[batch[n] * D_ + c0];
  atomicAdd(p + 0, v.x);
  atomicAdd(p + 1, v.y);
  atomicAdd(p + 2, v.z);
  atomicAdd(p + 3, v.w);
}

__global__ void k_bn_stats(const float* __restrict__ pooled,
                           float* __restrict__ stats) {
  const int c = threadIdx.x;  // 128 threads, one block
  float s = 0.f, ss = 0.f;
  for (int g = 0; g < G_; ++g) {
    const float v = pooled[g * D_ + c];
    s += v; ss += v * v;
  }
  const float mu = s / (float)G_;
  stats[c]       = mu;
  stats[D_ + c]  = ss / (float)G_ - mu * mu;   // population variance (ddof=0)
}

__global__ void k_fc(const float* __restrict__ pooled,
                     const float* __restrict__ stats,
                     const float* __restrict__ gamma,
                     const float* __restrict__ beta,
                     const float* __restrict__ fcw,
                     const float* __restrict__ fcb,
                     float* __restrict__ out) {
  int i = blockIdx.x * blockDim.x + threadIdx.x;
  if (i >= G_ * LAT_) return;
  const int g = i >> 6, j = i & 63;
  float acc = fcb[j];
  for (int c = 0; c < D_; ++c) {
    const float nv = (pooled[g * D_ + c] - stats[c]) *
                     rsqrtf(stats[D_ + c] + BN_EPS) * gamma[c] + beta[c];
    acc += nv * fcw[j * D_ + c];
  }
  out[i] = acc;
}

// ---------------------------------------------------------------------------
extern "C" void kernel_launch(void* const* d_in, const int* in_sizes, int n_in,
                              void* d_out, int out_size, void* d_ws,
                              size_t ws_size, hipStream_t stream) {
  const float* x      = (const float*)d_in[0];
  const int*   ei     = (const int*)  d_in[1];
  const int*   batch  = (const int*)  d_in[2];
  const float* W      = (const float*)d_in[3];
  const float* attS   = (const float*)d_in[4];
  const float* attD   = (const float*)d_in[5];
  const float* bias   = (const float*)d_in[6];
  const float* gamma  = (const float*)d_in[7];
  const float* beta   = (const float*)d_in[8];
  const float* fcw    = (const float*)d_in[9];
  const float* fcb    = (const float*)d_in[10];
  float* out = (float*)d_out;

  // workspace carve-out (256B aligned slices)
  char* wp = (char*)d_ws;
  auto alloc = [&](size_t bytes) {
    void* p = wp; wp += (bytes + 255) & ~(size_t)255; return p;
  };
  float*    hA     = (float*)   alloc((size_t)N_ * D_ * 4);  // layer output
  float*    hB     = (float*)   alloc((size_t)N_ * D_ * 4);  // projection
  float*    asrc   = (float*)   alloc((size_t)N_ * H_ * 4);
  float*    adst   = (float*)   alloc((size_t)N_ * H_ * 4);
  unsigned* segmax = (unsigned*)alloc((size_t)N_ * H_ * 4);
  float*    segsum = (float*)   alloc((size_t)N_ * H_ * 4);
  float*    alphaE = (float*)   alloc((size_t)EP_ * H_ * 4);
  float*    pooled = (float*)   alloc((size_t)G_ * D_ * 4);
  float*    stats  = (float*)   alloc((size_t)2 * D_ * 4);
  __bf16*   Wt     = (__bf16*)  alloc((size_t)L_ * D_ * D_ * 2);

  const int TB = 256;
  const int gNH = (N_ * H_ + TB - 1) / TB;    // node x head
  const int gN4 = (N_ * 32 + TB - 1) / TB;    // node x feature/4
  const int gEH = (EP_ * H_ + TB - 1) / TB;   // edge x head
  const int gE4 = (EP_ * 32 + TB - 1) / TB;   // edge x feature/4

  k_w_to_bf16t<<<(L_ * D_ * D_ + TB - 1) / TB, TB, 0, stream>>>(W, Wt);

  const float* feat = x;
  for (int l = 0; l < L_; ++l) {
    k_gemm_wmma<<<N_ / ROWS_, TB, 0, stream>>>(feat, Wt + (size_t)l * D_ * D_,
                                               hB);
    k_node_alpha<<<gNH, TB, 0, stream>>>(hB, attS + l * H_ * C_,
                                         attD + l * H_ * C_, asrc, adst);
    k_seg_init<<<gNH, TB, 0, stream>>>(segmax, segsum);
    k_edge_max<<<gEH, TB, 0, stream>>>(ei, asrc, adst, segmax);
    k_edge_sum<<<gEH, TB, 0, stream>>>(ei, asrc, adst, segmax, segsum);
    k_edge_alpha<<<gEH, TB, 0, stream>>>(ei, asrc, adst, segmax, segsum, alphaE);
    k_out_init<<<gN4, TB, 0, stream>>>((float4*)hA, (const float4*)bias + l * 32);
    k_aggregate<<<gE4, TB, 0, stream>>>(ei, hB, alphaE, hA);
    k_elu<<<gN4, TB, 0, stream>>>((float4*)hA);
    feat = hA;
  }

  k_pool_init<<<(G_ * D_ + TB - 1) / TB, TB, 0, stream>>>(pooled);
  k_pool<<<gN4, TB, 0, stream>>>(hA, batch, pooled);
  k_bn_stats<<<1, D_, 0, stream>>>(pooled, stats);
  k_fc<<<(G_ * LAT_ + TB - 1) / TB, TB, 0, stream>>>(pooled, stats, gamma,
                                                     beta, fcw, fcb, out);
}
